// My_GATRNNConv_25572235280998
// MI455X (gfx1250) — matrix operand order repaired and checked by reading, not compiled, not run on hardware
//
#include <hip/hip_runtime.h>
#include <hip/hip_bf16.h>
#include <math.h>

typedef __bf16 bf16;
typedef __attribute__((ext_vector_type(16))) __bf16 v16bf;
typedef __attribute__((ext_vector_type(8)))  __bf16 v8bf;
typedef __attribute__((ext_vector_type(8)))  float  v8f;

#define HEADS 8
#define CH    128
#define HC    1024   // HEADS*CH
#define NEG_SLOPE 0.2f

// ---------- order-preserving float<->uint encoding for atomicMax ----------
__device__ __forceinline__ unsigned enc_f32(float f) {
    unsigned u = __float_as_uint(f);
    return u ^ ((unsigned)((int)u >> 31) | 0x80000000u);
}
__device__ __forceinline__ float dec_f32(unsigned u) {
    unsigned s = (u & 0x80000000u) ? (u ^ 0x80000000u) : ~u;
    return __uint_as_float(s);
}

// ---------- small helpers ----------
__global__ void k_cast_bf16(const float* __restrict__ s, bf16* __restrict__ d, size_t n) {
    size_t i = (size_t)blockIdx.x * blockDim.x + threadIdx.x;
    if (i < n) d[i] = (bf16)s[i];
}

// W is [R, Ccols] row-major fp32 -> Bt is [Ccols, R] row-major bf16 (i.e. W^T)
__global__ void k_transpose_cast(const float* __restrict__ W, bf16* __restrict__ Bt,
                                 int R, int Ccols) {
    int i = blockIdx.x * blockDim.x + threadIdx.x;
    if (i >= R * Ccols) return;
    int r = i / Ccols, c = i % Ccols;
    Bt[(size_t)c * R + r] = (bf16)W[i];
}

// ---------- bf16 WMMA GEMM: C[M,N] = act(A[M,K] @ B[K,N] + bias) ----------
// Block = 256 threads = 8 waves. One block owns 8 consecutive 16-row tiles of
// a single 16-column panel. The K x 16 B panel is staged in LDS in a
// fragment-major swizzle so each lane's 16 B elements are 32 contiguous bytes:
//   swizzle(k, c) = (k>>5)*512 + ((k>>3)&1)*256 + c*16 + ((k>>4)&1)*8 + (k&7)
// => B fragment = one 32B LDS read (2x ds_load_b128) per K-step.
// All output-mode decisions are compile-time: branch-free epilogue.
template<int K, bool WF32, bool WBF16, bool HAS_BIAS, bool RELU>
__global__ void k_wmma_gemm(const bf16* __restrict__ A, const bf16* __restrict__ B,
                            const float* __restrict__ bias,
                            float* __restrict__ C, bf16* __restrict__ Cbf,
                            int M, int N)
{
    __shared__ __align__(32) bf16 ldsB[K * 16];

    int lane  = threadIdx.x & 31;
    int wave  = threadIdx.x >> 5;
    int tilesN = N >> 4;
    int tilesM = M >> 4;
    int bm = blockIdx.x / tilesN;                    // group of 8 row-tiles
    int tn = blockIdx.x % tilesN;                    // column tile
    int tm = bm * 8 + wave;

    // cooperative fill, walking the swizzled index linearly (contiguous LDS writes)
    for (int sw = threadIdx.x; sw < K * 16; sw += 256) {
        int e    = sw & 15;
        int c    = (sw >> 4) & 15;
        int hl2  = (sw >> 8) & 1;
        int kblk = sw >> 9;
        int k    = (kblk << 5) + ((e >> 3) << 4) + (hl2 << 3) + (e & 7);
        ldsB[sw] = B[(size_t)k * N + tn * 16 + c];
    }
    __syncthreads();

    if (tm >= tilesM) return;                        // wave-uniform: EXEC stays full

    int lh = lane >> 4;                              // lane half (0: lanes 0-15)
    int lr = lane & 15;
    const bf16* Arow = A + (size_t)(tm * 16 + lr) * K;

    v8f acc = {};
    #pragma unroll 4
    for (int kb = 0; kb < K; kb += 32) {
        // A 16x32 fragment: e=0..7 -> K = kb+lh*8+e ; e=8..15 -> +16
        const bf16* ap = Arow + kb + lh * 8;
        v8bf alo = *(const v8bf*)ap;                 // 16B global load
        v8bf ahi = *(const v8bf*)(ap + 16);          // 16B global load
        v16bf a;
        #pragma unroll
        for (int i = 0; i < 8; ++i) { a[i] = alo[i]; a[8 + i] = ahi[i]; }

        // B 32x16 fragment: 32 contiguous bytes in swizzled LDS
        v16bf b = *(const v16bf*)(ldsB + (kb << 4) + (lh << 8) + (lr << 4));

        acc = __builtin_amdgcn_wmma_f32_16x16x32_bf16(false, a, false, b,
                                                      (short)0, acc, false, false);
    }

    int col = tn * 16 + lr;
    float bv = HAS_BIAS ? bias[col] : 0.0f;
    #pragma unroll
    for (int v = 0; v < 8; ++v) {
        int row = tm * 16 + lh * 8 + v;              // C/D: VGPR v -> M = v + 8*lh
        float val = acc[v] + bv;
        if constexpr (RELU) val = fmaxf(val, 0.0f);
        size_t idx = (size_t)row * N + col;
        if constexpr (WF32)  C[idx]   = val;
        if constexpr (WBF16) Cbf[idx] = (bf16)val;
    }
}

// ---------- per-(node,head) attention scores ----------
__global__ void k_att_scores(const float* __restrict__ h, const float* __restrict__ att_s,
                             const float* __restrict__ att_d, float* __restrict__ asrc,
                             float* __restrict__ adst, int Nn)
{
    int t = blockIdx.x * blockDim.x + threadIdx.x;
    if (t >= Nn * HEADS) return;
    int n = t >> 3, hd = t & 7;
    const float* hp = h + (size_t)n * HC + hd * CH;
    const float* as = att_s + hd * CH;
    const float* ad = att_d + hd * CH;
    float s1 = 0.f, s2 = 0.f;
    for (int c = 0; c < CH; ++c) { float v = hp[c]; s1 += v * as[c]; s2 += v * ad[c]; }
    asrc[t] = s1; adst[t] = s2;
}

// ---------- edge pass 1: leaky-relu logits + segment max ----------
__global__ void k_edge_logits(const int* __restrict__ ei, const float* __restrict__ asrc,
                              const float* __restrict__ adst, float* __restrict__ logits,
                              unsigned* __restrict__ amax, int E, int Etot)
{
    int t = blockIdx.x * blockDim.x + threadIdx.x;
    if (t >= Etot * HEADS) return;
    int e = t >> 3, hd = t & 7;
    int s = (e < E) ? ei[e]     : (e - E);
    int d = (e < E) ? ei[E + e] : (e - E);
    float v = asrc[s * HEADS + hd] + adst[d * HEADS + hd];
    v = (v > 0.f) ? v : NEG_SLOPE * v;
    logits[t] = v;
    atomicMax(&amax[d * HEADS + hd], enc_f32(v));
}

// ---------- edge pass 2: exp + segment sum ----------
__global__ void k_edge_exp(const int* __restrict__ ei, float* logits,
                           const unsigned* __restrict__ amax, float* __restrict__ denom,
                           int E, int Etot)
{
    int t = blockIdx.x * blockDim.x + threadIdx.x;
    if (t >= Etot * HEADS) return;
    int e = t >> 3, hd = t & 7;
    int d = (e < E) ? ei[E + e] : (e - E);
    float a = __expf(logits[t] - dec_f32(amax[d * HEADS + hd]));
    logits[t] = a;
    atomicAdd(&denom[d * HEADS + hd], a);
}

// ---------- edge pass 3: normalized weighted scatter; 1 wave per (edge,head) ----------
__global__ void k_edge_aggregate(const int* __restrict__ ei, const float* __restrict__ logits,
                                 const float* __restrict__ denom, const float* __restrict__ h,
                                 float* __restrict__ out, int E, int Etot)
{
    int lane = threadIdx.x & 31;
    int wave = threadIdx.x >> 5;
    long long wid = (long long)blockIdx.x * 8 + wave;
    if (wid >= (long long)Etot * HEADS) return;
    int e  = (int)(wid >> 3), hd = (int)(wid & 7);
    int s = (e < E) ? ei[e]     : (e - E);
    int d = (e < E) ? ei[E + e] : (e - E);
    float alpha = logits[wid] / (denom[d * HEADS + hd] + 1e-16f);
    const float* hs = h   + (size_t)s * HC + hd * CH;
    float*       od = out + (size_t)d * HC + hd * CH;
    #pragma unroll
    for (int i = 0; i < 4; ++i) {
        int c = lane + 32 * i;
        atomicAdd(&od[c], alpha * hs[c]);
    }
}

// ---------- m = bf16(relu(agg + b_gat)) ----------
__global__ void k_bias_relu_cast(const float* __restrict__ in, const float* __restrict__ bias,
                                 bf16* __restrict__ outb, size_t n, int modc)
{
    size_t i = (size_t)blockIdx.x * blockDim.x + threadIdx.x;
    if (i >= n) return;
    float v = in[i] + bias[i % modc];
    outb[i] = (bf16)fmaxf(v, 0.f);
}

// ---------- GRU cell elementwise + tanh; emit bf16 for final GEMM ----------
__global__ void k_gru(const float* __restrict__ gi, const float* __restrict__ gh,
                      const float* __restrict__ x, bf16* __restrict__ x2b, int Nn)
{
    int t = blockIdx.x * blockDim.x + threadIdx.x;
    if (t >= Nn * CH) return;
    int n = t >> 7, c = t & 127;
    const float* gir = gi + (size_t)n * 3 * CH;
    const float* ghr = gh + (size_t)n * 3 * CH;
    float r  = 1.f / (1.f + __expf(-(gir[c]        + ghr[c])));
    float z  = 1.f / (1.f + __expf(-(gir[CH + c]   + ghr[CH + c])));
    float nc = tanhf(gir[2 * CH + c] + r * ghr[2 * CH + c]);
    float hn = (1.f - z) * nc + z * x[t];
    x2b[t] = (bf16)tanhf(hn);
}

extern "C" void kernel_launch(void* const* d_in, const int* in_sizes, int n_in,
                              void* d_out, int out_size, void* d_ws, size_t ws_size,
                              hipStream_t stream) {
    (void)n_in; (void)out_size; (void)ws_size;
    const float* x       = (const float*)d_in[0];
    const int*   ei      = (const int*)  d_in[1];
    const float* W_gat   = (const float*)d_in[2];
    const float* att_src = (const float*)d_in[3];
    const float* att_dst = (const float*)d_in[4];
    const float* b_gat   = (const float*)d_in[5];
    const float* W_c     = (const float*)d_in[6];
    const float* b_c     = (const float*)d_in[7];
    const float* W_ih    = (const float*)d_in[8];
    const float* W_hh    = (const float*)d_in[9];
    const float* b_ih    = (const float*)d_in[10];
    const float* b_hh    = (const float*)d_in[11];
    const float* W_o     = (const float*)d_in[12];
    const float* b_o     = (const float*)d_in[13];

    int Nn   = in_sizes[0] / CH;   // 30000
    int E    = in_sizes[1] / 2;    // 480000
    int Etot = E + Nn;

    // ---- workspace carve ----
    char* p = (char*)d_ws;
    auto carve = [&](size_t bytes) -> char* {
        char* r = p; p += (bytes + 255) & ~(size_t)255; return r;
    };
    bf16*     x_bf    = (bf16*)    carve((size_t)Nn * CH * 2);
    bf16*     Wgat_bf = (bf16*)    carve((size_t)CH * HC * 2);
    bf16*     WcT     = (bf16*)    carve((size_t)HC * CH * 2);
    bf16*     WihT    = (bf16*)    carve((size_t)CH * 3 * CH * 2);
    bf16*     WhhT    = (bf16*)    carve((size_t)CH * 3 * CH * 2);
    bf16*     WoT     = (bf16*)    carve((size_t)CH * CH * 2);
    float*    h       = (float*)   carve((size_t)Nn * HC * 4);
    float*    asrc    = (float*)   carve((size_t)Nn * HEADS * 4);
    float*    adst    = (float*)   carve((size_t)Nn * HEADS * 4);
    unsigned* amax    = (unsigned*)carve((size_t)Nn * HEADS * 4);
    float*    denom   = (float*)   carve((size_t)Nn * HEADS * 4);
    float*    logits  = (float*)   carve((size_t)Etot * HEADS * 4);
    float*    agg     = (float*)   carve((size_t)Nn * HC * 4);
    bf16*     m_bf    = (bf16*)    carve((size_t)Nn * HC * 2);
    bf16*     mc_bf   = (bf16*)    carve((size_t)Nn * CH * 2);
    float*    gi      = (float*)   carve((size_t)Nn * 3 * CH * 4);
    float*    gh      = (float*)   carve((size_t)Nn * 3 * CH * 4);
    bf16*     x2_bf   = (bf16*)    carve((size_t)Nn * CH * 2);

    const int T = 256;
    // ---- casts / weight transposes (tiny) ----
    k_cast_bf16<<<(unsigned)(((size_t)Nn * CH + T - 1) / T), T, 0, stream>>>(x, x_bf, (size_t)Nn * CH);
    k_cast_bf16<<<(CH * HC + T - 1) / T, T, 0, stream>>>(W_gat, Wgat_bf, (size_t)CH * HC);
    k_transpose_cast<<<(CH * HC + T - 1) / T, T, 0, stream>>>(W_c, WcT, CH, HC);
    k_transpose_cast<<<(3 * CH * CH + T - 1) / T, T, 0, stream>>>(W_ih, WihT, 3 * CH, CH);
    k_transpose_cast<<<(3 * CH * CH + T - 1) / T, T, 0, stream>>>(W_hh, WhhT, 3 * CH, CH);
    k_transpose_cast<<<(CH * CH + T - 1) / T, T, 0, stream>>>(W_o, WoT, CH, CH);

    int tilesM = Nn / 16;                       // 1875
    int bmCnt  = (tilesM + 7) / 8;              // 8 row-tiles per block

    // ---- h = x @ W_gat  (M=Nn, K=128, N=1024), fp32 out, no bias ----
    k_wmma_gemm<CH, true, false, false, false>
        <<<bmCnt * (HC / 16), 256, 0, stream>>>(x_bf, Wgat_bf, nullptr, h, nullptr, Nn, HC);

    k_att_scores<<<(Nn * HEADS + T - 1) / T, T, 0, stream>>>(h, att_src, att_dst, asrc, adst, Nn);

    hipMemsetAsync(amax,  0, (size_t)Nn * HEADS * 4, stream);
    hipMemsetAsync(denom, 0, (size_t)Nn * HEADS * 4, stream);
    hipMemsetAsync(agg,   0, (size_t)Nn * HC * 4,    stream);

    int edgeThreads = Etot * HEADS;
    k_edge_logits<<<(edgeThreads + T - 1) / T, T, 0, stream>>>(ei, asrc, adst, logits, amax, E, Etot);
    k_edge_exp   <<<(edgeThreads + T - 1) / T, T, 0, stream>>>(ei, logits, amax, denom, E, Etot);
    k_edge_aggregate<<<(unsigned)(((long long)Etot * HEADS + 7) / 8), 256, 0, stream>>>(
        ei, logits, denom, h, agg, E, Etot);

    k_bias_relu_cast<<<(unsigned)(((size_t)Nn * HC + T - 1) / T), T, 0, stream>>>(
        agg, b_gat, m_bf, (size_t)Nn * HC, HC);

    // ---- mc = relu(m @ W_c^T + b_c)  (M=Nn, K=1024, N=128) -> bf16 only ----
    k_wmma_gemm<HC, false, true, true, true>
        <<<bmCnt * (CH / 16), 256, 0, stream>>>(m_bf, WcT, b_c, nullptr, mc_bf, Nn, CH);

    // ---- GRU gate GEMMs (M=Nn, K=128, N=384), fp32 out + bias ----
    k_wmma_gemm<CH, true, false, true, false>
        <<<bmCnt * (3 * CH / 16), 256, 0, stream>>>(mc_bf, WihT, b_ih, gi, nullptr, Nn, 3 * CH);
    k_wmma_gemm<CH, true, false, true, false>
        <<<bmCnt * (3 * CH / 16), 256, 0, stream>>>(x_bf,  WhhT, b_hh, gh, nullptr, Nn, 3 * CH);

    k_gru<<<(Nn * CH + T - 1) / T, T, 0, stream>>>(gi, gh, x, x2_bf, Nn);

    // ---- out = x2 @ W_o^T + b_o  (M=Nn, K=128, N=128), fp32 out + bias ----
    k_wmma_gemm<CH, true, false, true, false>
        <<<bmCnt * (CH / 16), 256, 0, stream>>>(x2_bf, WoT, b_o, (float*)d_out, Nn ? nullptr : nullptr, Nn, CH);
}
